// UniversalHydraulicGNN_51943334478564
// MI455X (gfx1250) — compile-verified, matrix operand order checked by reading, not run on hardware
//
#include <hip/hip_runtime.h>

#define H_HEADS 4
#define C_DIM   96
#define D_DIM   384
#define NCLS    20
#define NEG_SLOPE 0.2f
#define LN_EPS  1e-5f

typedef __attribute__((ext_vector_type(16))) _Float16 v16h;
typedef __attribute__((ext_vector_type(8)))  float    v8f;

// ---------- fills ----------
__global__ void fill_u32(unsigned* p, unsigned v, int n) {
  int i = blockIdx.x * blockDim.x + threadIdx.x;
  if (i < n) p[i] = v;
}

// ---------- cast fp32 activations -> f16, pad K to Kpad with zeros ----------
__global__ void cast_pad_f16(const float* __restrict__ src, _Float16* __restrict__ dst,
                             int N, int K, int Kpad) {
  int idx = blockIdx.x * blockDim.x + threadIdx.x;
  if (idx >= N * Kpad) return;
  int i = idx / Kpad, k = idx % Kpad;
  dst[idx] = (k < K) ? (_Float16)src[(size_t)i * K + k] : (_Float16)0.f;
}

// ---------- transpose W [K x Ncol] fp32 -> Wt [Ncol x Kpad] f16 (zero pad) ----------
__global__ void transpose_w_f16(const float* __restrict__ W, _Float16* __restrict__ Wt,
                                int K, int Kpad, int Ncol) {
  int idx = blockIdx.x * blockDim.x + threadIdx.x;
  if (idx >= Ncol * Kpad) return;
  int n = idx / Kpad, k = idx % Kpad;
  Wt[idx] = (k < K) ? (_Float16)W[(size_t)k * Ncol + n] : (_Float16)0.f;
}

// ---------- WMMA GEMM: C[M x Ncol] = A[M x K] * B, with Bt[n][k] = B[k][n] ----------
// One wave computes a 16x32 output tile (A fragment reused over two WMMAs).
// Requires: M % 16 == 0, K % 32 == 0, Ncol % 32 == 0.
__global__ void gemm_wmma_f16(const _Float16* __restrict__ A,
                              const _Float16* __restrict__ Bt,
                              float* __restrict__ Cmat,
                              int M, int K, int Ncol, int tilesTotal) {
  const int wave = (blockIdx.x * blockDim.x + threadIdx.x) >> 5;
  if (wave >= tilesTotal) return;            // whole-wave exit: EXEC stays all-ones for WMMA
  const int lane  = threadIdx.x & 31;
  const int tilesN = Ncol >> 5;
  const int tm = wave / tilesN;
  const int tn = wave % tilesN;
  const int halfL = lane >> 4;               // 0: lanes 0-15, 1: lanes 16-31
  const int l16   = lane & 15;

  const _Float16* arow  = A  + (size_t)(tm * 16 + l16) * K;
  const _Float16* brow0 = Bt + (size_t)(tn * 32 + l16) * K;
  const _Float16* brow1 = brow0 + (size_t)16 * K;

  v8f acc0 = {};
  v8f acc1 = {};
  for (int k0 = 0; k0 < K; k0 += 32) {
    v16h a, b0, b1;
    const int kb = k0 + halfL * 8;           // ISA 16-bit A layout: lanes 16-31 start at K+8
#pragma unroll
    for (int v = 0; v < 4; ++v) {
      int klo = kb + 2 * v;                  // VGPR v   : K = klo, klo+1
      int khi = klo + 16;                    // VGPR v+4 : K = klo+16, klo+17
      a[2*v]    = arow[klo];   a[2*v+1]  = arow[klo+1];
      a[2*v+8]  = arow[khi];   a[2*v+9]  = arow[khi+1];
      b0[2*v]   = brow0[klo];  b0[2*v+1] = brow0[klo+1];
      b0[2*v+8] = brow0[khi];  b0[2*v+9] = brow0[khi+1];
      b1[2*v]   = brow1[klo];  b1[2*v+1] = brow1[klo+1];
      b1[2*v+8] = brow1[khi];  b1[2*v+9] = brow1[khi+1];
    }
    acc0 = __builtin_amdgcn_wmma_f32_16x16x32_f16(false, a, false, b0, (short)0, acc0, false, false);
    acc1 = __builtin_amdgcn_wmma_f32_16x16x32_f16(false, a, false, b1, (short)0, acc1, false, false);
  }
  // C/D layout: lane = N + (M>=8 ? 16 : 0), VGPR r holds M = r (+8 for upper lanes)
  float* c0 = Cmat + (size_t)(tm * 16 + halfL * 8) * Ncol + tn * 32 + l16;
#pragma unroll
  for (int r = 0; r < 8; ++r) {
    c0[(size_t)r * Ncol]      = acc0[r];
    c0[(size_t)r * Ncol + 16] = acc1[r];
  }
}

// ================= CSR build (once per launch; edge_index is constant) =================
__global__ void csr_count(const int* __restrict__ dst, int* __restrict__ deg,
                          int* __restrict__ pos, int E) {
  int e = blockIdx.x * blockDim.x + threadIdx.x;
  if (e >= E) return;
  pos[e] = atomicAdd(&deg[dst[e]], 1);
}

// exclusive scan, 256 elements per block (Hillis-Steele in LDS)
__global__ void scan_block(const int* __restrict__ in, int* __restrict__ out,
                           int* __restrict__ bsum, int n) {
  __shared__ int tmp[256];
  int tid = threadIdx.x;
  int i = blockIdx.x * 256 + tid;
  int v = (i < n) ? in[i] : 0;
  tmp[tid] = v;
  __syncthreads();
  for (int off = 1; off < 256; off <<= 1) {
    int t = (tid >= off) ? tmp[tid - off] : 0;
    __syncthreads();
    tmp[tid] += t;
    __syncthreads();
  }
  if (i < n) out[i] = tmp[tid] - v;          // exclusive
  if (tid == 255) bsum[blockIdx.x] = tmp[255];
}

// exclusive scan of block sums (single block; nb <= 256)
__global__ void scan_sums(int* __restrict__ bsum, int nb) {
  __shared__ int tmp[256];
  int tid = threadIdx.x;
  int v = (tid < nb) ? bsum[tid] : 0;
  tmp[tid] = v;
  __syncthreads();
  for (int off = 1; off < 256; off <<= 1) {
    int t = (tid >= off) ? tmp[tid - off] : 0;
    __syncthreads();
    tmp[tid] += t;
    __syncthreads();
  }
  if (tid < nb) bsum[tid] = tmp[tid] - v;
}

__global__ void scan_add(int* __restrict__ data, const int* __restrict__ bsum, int n) {
  int i = blockIdx.x * 256 + threadIdx.x;
  if (i < n) data[i] += bsum[blockIdx.x];
}

__global__ void csr_scatter(const int* __restrict__ dst, const int* __restrict__ rowstart,
                            const int* __restrict__ pos, int* __restrict__ eidx, int E) {
  int e = blockIdx.x * blockDim.x + threadIdx.x;
  if (e >= E) return;
  eidx[rowstart[dst[e]] + pos[e]] = e;
}

// ========== fused per-node GATv2: online softmax + aggregate + bias + LN + ELU ==========
// 1 block (128 threads = 4 wave32) per node; wave h handles head h; lane owns 3 channels.
__global__ void gat_node(const int* __restrict__ src,
                         const int* __restrict__ rowstart, const int* __restrict__ deg,
                         const int* __restrict__ eidx,
                         const float* __restrict__ xs, const float* __restrict__ xd,
                         const float* __restrict__ att,
                         const float* __restrict__ bias, const float* __restrict__ g,
                         const float* __restrict__ be,
                         float* __restrict__ hout, int N) {
  const int node = blockIdx.x;
  const int lane = threadIdx.x & 31;
  const int head = threadIdx.x >> 5;         // 0..3
  const int c0   = head * C_DIM + lane * 3;  // channel base within D=384

  const int base = rowstart[node];
  const int cnt  = deg[node];

  const float a0 = att[c0], a1 = att[c0 + 1], a2 = att[c0 + 2];
  const float* pdl = xd + (size_t)node * D_DIM + c0;
  const float d0 = pdl[0], d1 = pdl[1], d2 = pdl[2];

  float m = -INFINITY, l = 0.f;
  float acc0 = 0.f, acc1 = 0.f, acc2 = 0.f;

  for (int j = 0; j < cnt; ++j) {
    int e = eidx[base + j];
    int s = src[e];
    const float* ps = xs + (size_t)s * D_DIM + c0;
    float x0 = ps[0], x1 = ps[1], x2 = ps[2];
    float t0 = x0 + d0, t1 = x1 + d1, t2 = x2 + d2;
    t0 = (t0 > 0.f) ? t0 : NEG_SLOPE * t0;
    t1 = (t1 > 0.f) ? t1 : NEG_SLOPE * t1;
    t2 = (t2 > 0.f) ? t2 : NEG_SLOPE * t2;
    float part = fmaf(a0, t0, fmaf(a1, t1, a2 * t2));
    for (int o = 16; o > 0; o >>= 1) part += __shfl_down(part, o, 32);
    float ej = __shfl(part, 0, 32);
    float mn = fmaxf(m, ej);
    float scale = __expf(m - mn);            // first iter: exp(-inf) = 0
    float p = __expf(ej - mn);
    l = l * scale + p;
    acc0 = fmaf(acc0, scale, p * x0);
    acc1 = fmaf(acc1, scale, p * x1);
    acc2 = fmaf(acc2, scale, p * x2);
    m = mn;
  }

  const float inv = 1.f / (l + 1e-16f);
  float o0 = acc0 * inv + bias[c0];
  float o1 = acc1 * inv + bias[c0 + 1];
  float o2 = acc2 * inv + bias[c0 + 2];

  // block-wide LayerNorm over 384 channels (each thread owns 3)
  __shared__ float wsum[4];
  __shared__ float stat;
  float s = o0 + o1 + o2;
  for (int o = 16; o > 0; o >>= 1) s += __shfl_down(s, o, 32);
  if (lane == 0) wsum[head] = s;
  __syncthreads();
  if (threadIdx.x == 0) stat = (wsum[0] + wsum[1] + wsum[2] + wsum[3]) * (1.f / D_DIM);
  __syncthreads();
  float mu = stat;
  __syncthreads();
  s = (o0 - mu) * (o0 - mu) + (o1 - mu) * (o1 - mu) + (o2 - mu) * (o2 - mu);
  for (int o = 16; o > 0; o >>= 1) s += __shfl_down(s, o, 32);
  if (lane == 0) wsum[head] = s;
  __syncthreads();
  if (threadIdx.x == 0) stat = (wsum[0] + wsum[1] + wsum[2] + wsum[3]) * (1.f / D_DIM);
  __syncthreads();
  float rstd = rsqrtf(stat + LN_EPS);

  float* po = hout + (size_t)node * D_DIM + c0;
  float y0 = (o0 - mu) * rstd * g[c0]     + be[c0];
  float y1 = (o1 - mu) * rstd * g[c0 + 1] + be[c0 + 1];
  float y2 = (o2 - mu) * rstd * g[c0 + 2] + be[c0 + 2];
  po[0] = (y0 > 0.f) ? y0 : (__expf(y0) - 1.f);
  po[1] = (y1 > 0.f) ? y1 : (__expf(y1) - 1.f);
  po[2] = (y2 > 0.f) ? y2 : (__expf(y2) - 1.f);
}

// ---------- head: relu(hc + bc1) then layernorm, C=96, 96 threads ----------
__global__ void head_norm_96(float* __restrict__ hio, const float* __restrict__ b1,
                             const float* __restrict__ g, const float* __restrict__ be, int Nn) {
  int node = blockIdx.x;
  int t = threadIdx.x;
  float v = fmaxf(hio[(size_t)node * C_DIM + t] + b1[t], 0.f);
  __shared__ float wsum[3];
  __shared__ float stat;
  float s = v;
  for (int o = 16; o > 0; o >>= 1) s += __shfl_down(s, o, 32);
  if ((t & 31) == 0) wsum[t >> 5] = s;
  __syncthreads();
  if (t == 0) stat = (wsum[0] + wsum[1] + wsum[2]) * (1.f / C_DIM);
  __syncthreads();
  float mu = stat;
  __syncthreads();
  s = (v - mu) * (v - mu);
  for (int o = 16; o > 0; o >>= 1) s += __shfl_down(s, o, 32);
  if ((t & 31) == 0) wsum[t >> 5] = s;
  __syncthreads();
  if (t == 0) stat = (wsum[0] + wsum[1] + wsum[2]) * (1.f / C_DIM);
  __syncthreads();
  float var = stat;
  hio[(size_t)node * C_DIM + t] = (v - mu) * rsqrtf(var + LN_EPS) * g[t] + be[t];
}

// ---------- final fc: [N,96] x [96,20] + bias ----------
__global__ void final_fc(const float* __restrict__ hc, const float* __restrict__ W,
                         const float* __restrict__ b, float* __restrict__ out, int Nn) {
  int idx = blockIdx.x * blockDim.x + threadIdx.x;
  if (idx >= Nn * NCLS) return;
  int i = idx / NCLS, j = idx % NCLS;
  const float* hr = hc + (size_t)i * C_DIM;
  float acc = b[j];
#pragma unroll 8
  for (int c = 0; c < C_DIM; ++c) acc = fmaf(hr[c], W[c * NCLS + j], acc);
  out[idx] = acc;
}

extern "C" void kernel_launch(void* const* d_in, const int* in_sizes, int n_in,
                              void* d_out, int out_size, void* d_ws, size_t ws_size,
                              hipStream_t stream) {
  const int F_IN = 80;
  const int N = in_sizes[0] / F_IN;
  const int E = in_sizes[1] / 2;

  const float* x  = (const float*)d_in[0];
  const int*   ei = (const int*)d_in[1];
  const int*   src = ei;
  const int*   dst = ei + E;

  const float* Ws[3]  = {(const float*)d_in[2],  (const float*)d_in[8],  (const float*)d_in[14]};
  const float* Wd[3]  = {(const float*)d_in[3],  (const float*)d_in[9],  (const float*)d_in[15]};
  const float* att[3] = {(const float*)d_in[4],  (const float*)d_in[10], (const float*)d_in[16]};
  const float* bb[3]  = {(const float*)d_in[5],  (const float*)d_in[11], (const float*)d_in[17]};
  const float* gg[3]  = {(const float*)d_in[6],  (const float*)d_in[12], (const float*)d_in[18]};
  const float* be[3]  = {(const float*)d_in[7],  (const float*)d_in[13], (const float*)d_in[19]};
  const float* Wc1 = (const float*)d_in[20];
  const float* bc1 = (const float*)d_in[21];
  const float* gc  = (const float*)d_in[22];
  const float* bec = (const float*)d_in[23];
  const float* Wc2 = (const float*)d_in[24];
  const float* bc2 = (const float*)d_in[25];

  // workspace carve-out (~296 MB)
  char* wp = (char*)d_ws;
  auto take = [&](size_t bytes) { char* p = wp; wp += (bytes + 255) & ~(size_t)255; return p; };
  float*    h0   = (float*)take((size_t)N * D_DIM * 4);
  float*    h1   = (float*)take((size_t)N * D_DIM * 4);
  float*    xs   = (float*)take((size_t)N * D_DIM * 4);
  float*    xd   = (float*)take((size_t)N * D_DIM * 4);
  _Float16* xh   = (_Float16*)take((size_t)N * D_DIM * 2);
  _Float16* wt0  = (_Float16*)take((size_t)D_DIM * D_DIM * 2);
  _Float16* wt1  = (_Float16*)take((size_t)D_DIM * D_DIM * 2);
  int*      deg  = (int*)take((size_t)N * 4);
  int*      pos  = (int*)take((size_t)E * 4);
  int*      rowstart = (int*)take((size_t)N * 4);
  int*      eidx = (int*)take((size_t)E * 4);
  int*      bsum = (int*)take(256 * 4);
  float*    hc   = xs;   // head activations reuse xs (free by then)

  // ---- build dst-CSR once (reused by all 3 layers) ----
  const int nScanBlocks = (N + 255) / 256;   // 196 for N=50000 (fits single-block pass 2)
  fill_u32<<<(N + 255) / 256, 256, 0, stream>>>((unsigned*)deg, 0u, N);
  csr_count<<<(E + 255) / 256, 256, 0, stream>>>(dst, deg, pos, E);
  scan_block<<<nScanBlocks, 256, 0, stream>>>(deg, rowstart, bsum, N);
  scan_sums<<<1, 256, 0, stream>>>(bsum, nScanBlocks);
  scan_add<<<nScanBlocks, 256, 0, stream>>>(rowstart, bsum, N);
  csr_scatter<<<(E + 255) / 256, 256, 0, stream>>>(dst, rowstart, pos, eidx, E);

  float* hbuf[2] = {h0, h1};
  const float* hin = x;

  for (int l = 0; l < 3; ++l) {
    const int Kin  = (l == 0) ? F_IN : D_DIM;
    const int Kpad = (l == 0) ? 96   : D_DIM;
    float* hout = hbuf[l & 1];  // l0->h0 (reads x), l1->h1 (reads h0), l2->h0 (reads h1)

    { int n = N * Kpad;
      cast_pad_f16<<<(n + 255) / 256, 256, 0, stream>>>(hin, xh, N, Kin, Kpad); }
    { int n = D_DIM * Kpad;
      transpose_w_f16<<<(n + 255) / 256, 256, 0, stream>>>(Ws[l], wt0, Kin, Kpad, D_DIM);
      transpose_w_f16<<<(n + 255) / 256, 256, 0, stream>>>(Wd[l], wt1, Kin, Kpad, D_DIM); }
    { int tiles = (N / 16) * (D_DIM / 32);
      int blocks = (tiles + 3) / 4;   // 4 waves / 128-thread block
      gemm_wmma_f16<<<blocks, 128, 0, stream>>>(xh, wt0, xs, N, Kpad, D_DIM, tiles);
      gemm_wmma_f16<<<blocks, 128, 0, stream>>>(xh, wt1, xd, N, Kpad, D_DIM, tiles); }

    gat_node<<<N, 128, 0, stream>>>(src, rowstart, deg, eidx, xs, xd, att[l],
                                    bb[l], gg[l], be[l], hout, N);
    hin = hout;
  }

  // classifier head
  { int n = N * D_DIM;
    cast_pad_f16<<<(n + 255) / 256, 256, 0, stream>>>(hin, xh, N, D_DIM, D_DIM); }
  { int n = C_DIM * D_DIM;
    transpose_w_f16<<<(n + 255) / 256, 256, 0, stream>>>(Wc1, wt0, D_DIM, D_DIM, C_DIM); }
  { int tiles = (N / 16) * (C_DIM / 32);
    int blocks = (tiles + 3) / 4;
    gemm_wmma_f16<<<blocks, 128, 0, stream>>>(xh, wt0, hc, N, D_DIM, C_DIM, tiles); }
  head_norm_96<<<N, 96, 0, stream>>>(hc, bc1, gc, bec, N);
  { int n = N * NCLS;
    final_fc<<<(n + 255) / 256, 256, 0, stream>>>(hc, Wc2, bc2, (float*)d_out, N); }
}
